// LSTM_28492813041694
// MI455X (gfx1250) — matrix operand order repaired and checked by reading, not compiled, not run on hardware
//
#include <hip/hip_runtime.h>
#include <hip/hip_bf16.h>

// LSTM: B=256, T=512, I=64, H=512, O=1
// gates = [h | x_t] * [W_hh | W_ih]^T + (b_ih + b_hh),   K = H + I = 576
#define B_ 256
#define T_ 512
#define I_ 64
#define H_ 512
#define NBLK 64      // 32 j-groups x 2 batch halves
#define NKT  18      // 576 / 32 K-tiles
#define NNT  4       // 4 gate N-tiles (i,f,g,o) per j-group

typedef __attribute__((ext_vector_type(16))) __bf16 v16bf;
typedef __attribute__((ext_vector_type(8)))  float  v8f;

union BF16x16 {
  v16bf v;
  uint4 q[2];
  unsigned short us[16];
};

__device__ __forceinline__ unsigned short f2bf(float f) {
  unsigned u = __builtin_bit_cast(unsigned, f);
  u += 0x7FFFu + ((u >> 16) & 1u);           // round-to-nearest-even
  return (unsigned short)(u >> 16);
}
__device__ __forceinline__ float bf2f(unsigned short s) {
  return __builtin_bit_cast(float, ((unsigned)s) << 16);
}
__device__ __forceinline__ float sigmoidf_(float x) {
  return 1.0f / (1.0f + __expf(-x));
}
__device__ __forceinline__ float tanhf_(float x) {
  float e = __expf(-2.0f * x);
  return (1.0f - e) / (1.0f + e);
}

// ---------------------------------------------------------------- init kernel
// Zero the grid-barrier state and the t=0 hidden-state buffer every launch
// (d_ws is poisoned once and never re-poisoned -> we must re-init ourselves).
__global__ void lstm_init_kernel(unsigned* __restrict__ bar,
                                 unsigned* __restrict__ h0_u32) {
  int i = blockIdx.x * blockDim.x + threadIdx.x;
  if (i < 2) bar[i] = 0u;
  if (i < (B_ * H_ / 2)) h0_u32[i] = 0u;     // 256*512 bf16 == 65536 dwords
}

// ------------------------------------------------------------- grid barrier
__device__ __forceinline__ void grid_barrier(unsigned* bar) {
  __threadfence();                            // flush this wave's h stores
  __syncthreads();
  if (threadIdx.x == 0) {
    unsigned gen = atomicAdd(&bar[1], 0u);
    if (atomicAdd(&bar[0], 1u) == NBLK - 1u) {
      atomicExch(&bar[0], 0u);
      atomicAdd(&bar[1], 1u);                 // release everyone
    } else {
      while (atomicAdd(&bar[1], 0u) == gen) __builtin_amdgcn_s_sleep(1);
    }
  }
  __syncthreads();
}

// --------------------------------------------------------- persistent LSTM
// Block = 256 threads (8 wave32). blockIdx: [jgroup(32) x bhalf(2)].
// Wave w owns batch M-tile (bhalf*128 + w*16) and all 4 gate N-tiles of its
// 16 j columns -> i/f/g/o for one (b,j) land in the same lane & VGPR slot,
// so the cell update + c-state stay entirely in registers.
__global__ void __launch_bounds__(256)
lstm_persistent(const float* __restrict__ x,
                const float* __restrict__ W_ih,
                const float* __restrict__ W_hh,
                const float* __restrict__ b_ih,
                const float* __restrict__ b_hh,
                unsigned short* __restrict__ h0,
                unsigned short* __restrict__ h1,
                unsigned* __restrict__ bar) {
  extern __shared__ unsigned short Wlds[];   // [NNT][NKT][32 lanes][16 halves]

  const int tid    = threadIdx.x;
  const int lane   = tid & 31;
  const int wave   = tid >> 5;
  const int jgroup = blockIdx.x >> 1;
  const int bhalf  = blockIdx.x & 1;
  const int jbase  = jgroup * 16;
  const int bbase  = bhalf * 128 + wave * 16;

  // ---- Stage fused weights into LDS, pre-swizzled into WMMA B-operand
  //      per-lane order (B 32x16 bf16: halves 0-7 -> K kb..kb+7, halves 8-15
  //      -> K kb+16..kb+23, kb=8 for lanes 16-31, N = lane&15).  72 KB, resident for all 512 steps.
  for (int idx = tid; idx < NNT * NKT * 32 * 16; idx += 256) {
    int hh   = idx & 15;
    int ln   = (idx >> 4) & 31;
    int rest = idx >> 9;                     // nt*NKT + kt
    int kt   = rest % NKT;
    int nt   = rest / NKT;
    int k    = kt * 32 + ((ln >= 16) ? 8 : 0) + ((hh < 8) ? hh : hh + 8);
    int grow = nt * H_ + jbase + (ln & 15);  // gate nt, column j
    float w  = (k < H_) ? W_hh[(size_t)grow * H_ + k]
                        : W_ih[(size_t)grow * I_ + (k - H_)];
    Wlds[idx] = f2bf(w);
  }
  __syncthreads();

  // ---- per-lane gate biases (b_ih + b_hh), one per N-tile (= gate)
  float bias[NNT];
#pragma unroll
  for (int nt = 0; nt < NNT; ++nt) {
    int gi = nt * H_ + jbase + (lane & 15);
    bias[nt] = b_ih[gi] + b_hh[gi];
  }

  // ---- A-fragment addressing (A 16x32 bf16 per ISA: lanes 0-15 row M=lane,
  //      halves 0-7 -> K kb..kb+7, halves 8-15 -> K kb+16..kb+23, kb=8 for
  //      lanes 16-31)
  const int rowA = bbase + (lane & 15);
  const int kb   = (lane >= 16) ? 8 : 0;
  const int mrow = (lane >= 16) ? 8 : 0;

  float c_reg[8] = {0.f, 0.f, 0.f, 0.f, 0.f, 0.f, 0.f, 0.f};
  const v8f vzero = {0.f, 0.f, 0.f, 0.f, 0.f, 0.f, 0.f, 0.f};

#pragma unroll 1
  for (int t = 0; t < T_; ++t) {
    const unsigned short* hcur = (t & 1) ? h1 : h0;
    unsigned short*       hnxt = (t & 1) ? h0 : h1;

    v8f acc[NNT];
#pragma unroll
    for (int nt = 0; nt < NNT; ++nt) acc[nt] = vzero;

    // ---- recurrent K-tiles (k = 0..511) from bf16 h buffer
    const unsigned short* hbase = hcur + (size_t)rowA * H_ + kb;
#pragma unroll 4
    for (int kt = 0; kt < 16; ++kt) {
      BF16x16 a;
      a.q[0] = *reinterpret_cast<const uint4*>(hbase + kt * 32);
      a.q[1] = *reinterpret_cast<const uint4*>(hbase + kt * 32 + 16);
      if (kt < 14)  // WGP-scope prefetch of next-next K-tile (locality=3)
        __builtin_prefetch(hbase + (kt + 2) * 32, 0, 3);

      // Load ALL four gate B fragments into distinct registers first so the
      // compiler can clause the 8 ds_load_b128s, use partial s_wait_dscnt,
      // and avoid WAR stalls between a WMMA's B read and the next ds_load.
      BF16x16 bm[NNT];
#pragma unroll
      for (int nt = 0; nt < NNT; ++nt) {
        const unsigned short* wp = &Wlds[((nt * NKT + kt) * 32 + lane) * 16];
        bm[nt].q[0] = *reinterpret_cast<const uint4*>(wp);
        bm[nt].q[1] = *reinterpret_cast<const uint4*>(wp + 8);
      }
#pragma unroll
      for (int nt = 0; nt < NNT; ++nt) {
        acc[nt] = __builtin_amdgcn_wmma_f32_16x16x32_bf16(
            false, a.v, false, bm[nt].v, (short)0, acc[nt], false, false);
      }
    }

    // ---- input K-tiles (k = 512..575) from fp32 x, converted in-register
    const float* xrow = x + ((size_t)rowA * T_ + t) * I_ + kb;
#pragma unroll
    for (int kx = 0; kx < 2; ++kx) {
      BF16x16 a;
      const float* xp = xrow + kx * 32;
#pragma unroll
      for (int hh = 0; hh < 8; ++hh) {
        a.us[hh]     = f2bf(xp[hh]);
        a.us[8 + hh] = f2bf(xp[16 + hh]);
      }
      int kt = 16 + kx;
      BF16x16 bm[NNT];
#pragma unroll
      for (int nt = 0; nt < NNT; ++nt) {
        const unsigned short* wp = &Wlds[((nt * NKT + kt) * 32 + lane) * 16];
        bm[nt].q[0] = *reinterpret_cast<const uint4*>(wp);
        bm[nt].q[1] = *reinterpret_cast<const uint4*>(wp + 8);
      }
#pragma unroll
      for (int nt = 0; nt < NNT; ++nt) {
        acc[nt] = __builtin_amdgcn_wmma_f32_16x16x32_bf16(
            false, a.v, false, bm[nt].v, (short)0, acc[nt], false, false);
      }
    }

    // ---- in-register LSTM cell update (C/D layout: elem r -> M = mrow+r,
    //      N = lane&15), write h_{t+1} as bf16 for next step's A fragments
#pragma unroll
    for (int r = 0; r < 8; ++r) {
      float ig = sigmoidf_(acc[0][r] + bias[0]);
      float fg = sigmoidf_(acc[1][r] + bias[1]);
      float gg = tanhf_   (acc[2][r] + bias[2]);
      float og = sigmoidf_(acc[3][r] + bias[3]);
      float c  = fg * c_reg[r] + ig * gg;
      c_reg[r] = c;
      float hv = og * tanhf_(c);
      hnxt[(size_t)(bbase + mrow + r) * H_ + jbase + (lane & 15)] = f2bf(hv);
    }

    grid_barrier(bar);
  }
}

// ------------------------------------------------------------ output head
// out[b] = h_last[b,:] . W_out + b_out   (trivial: 256 dots of length 512)
__global__ void lstm_out_kernel(const unsigned short* __restrict__ h,
                                const float* __restrict__ W_out,
                                const float* __restrict__ b_out,
                                float* __restrict__ out) {
  int b = threadIdx.x;
  float s = 0.f;
  for (int j = 0; j < H_; ++j) s += bf2f(h[b * H_ + j]) * W_out[j];
  out[b] = s + b_out[0];
}

extern "C" void kernel_launch(void* const* d_in, const int* in_sizes, int n_in,
                              void* d_out, int out_size, void* d_ws, size_t ws_size,
                              hipStream_t stream) {
  const float* x     = (const float*)d_in[0];
  const float* W_ih  = (const float*)d_in[1];
  const float* W_hh  = (const float*)d_in[2];
  const float* b_ih  = (const float*)d_in[3];
  const float* b_hh  = (const float*)d_in[4];
  const float* W_out = (const float*)d_in[5];
  const float* b_out = (const float*)d_in[6];
  float* out = (float*)d_out;

  // workspace layout: [0,256) barrier, then two bf16 h buffers (256 KB each)
  unsigned*       bar = (unsigned*)d_ws;
  unsigned short* h0  = (unsigned short*)((char*)d_ws + 256);
  unsigned short* h1  = h0 + (size_t)B_ * H_;

  hipLaunchKernelGGL(lstm_init_kernel, dim3(256), dim3(256), 0, stream,
                     bar, (unsigned*)h0);

  const size_t smem = (size_t)NNT * NKT * 32 * 16 * sizeof(unsigned short); // 73728 B
  hipLaunchKernelGGL(lstm_persistent, dim3(NBLK), dim3(256), smem, stream,
                     x, W_ih, W_hh, b_ih, b_hh, h0, h1, bar);

  // T=512 even -> final hidden state is in h0
  hipLaunchKernelGGL(lstm_out_kernel, dim3(1), dim3(256), 0, stream,
                     h0, W_out, b_out, out);
}
